// FlashAttentionApprox_87445534146714
// MI455X (gfx1250) — matrix operand order repaired and checked by reading, not compiled, not run on hardware
//
#include <hip/hip_runtime.h>

typedef _Float16 v16h __attribute__((ext_vector_type(16)));
typedef float    v8f  __attribute__((ext_vector_type(8)));
typedef _Float16 h8   __attribute__((ext_vector_type(8)));
typedef _Float16 h4   __attribute__((ext_vector_type(4)));
typedef __fp16   pk2  __attribute__((ext_vector_type(2)));   // cvt_pkrtz result type

#define TDIM 2048
#define DDIM 64
#define BHDIM 32
#define BM 128            // query rows per workgroup (8 waves x 16)
#define BN 64             // keys per tile
#define NWAVE (BM / 16)
#define NTHR (NWAVE * 32)
#define LSTR 88           // LDS row stride in halves: 176B = 16B*11 (aligned, odd dword spread)
#define LITER ((BN * DDIM) / (NTHR * 4))   // staging iterations per tile (= 4)

// Reduce across the 16-lane group (lanes 0-15 / 16-31 independently) with DPP.
// After xor1+xor2 each 4-group is uniform, so half-mirror (i^7) == xor4 and
// row-mirror (i^15) == xor8 on the reduced values.
__device__ __forceinline__ float red_max16(float t) {
#if __has_builtin(__builtin_amdgcn_update_dpp)
    int x;
    x = __builtin_amdgcn_update_dpp(0, __builtin_bit_cast(int, t), 0xB1,  0xF, 0xF, true); // quad_perm [1,0,3,2]
    t = fmaxf(t, __builtin_bit_cast(float, x));
    x = __builtin_amdgcn_update_dpp(0, __builtin_bit_cast(int, t), 0x4E,  0xF, 0xF, true); // quad_perm [2,3,0,1]
    t = fmaxf(t, __builtin_bit_cast(float, x));
    x = __builtin_amdgcn_update_dpp(0, __builtin_bit_cast(int, t), 0x141, 0xF, 0xF, true); // row_half_mirror
    t = fmaxf(t, __builtin_bit_cast(float, x));
    x = __builtin_amdgcn_update_dpp(0, __builtin_bit_cast(int, t), 0x140, 0xF, 0xF, true); // row_mirror
    t = fmaxf(t, __builtin_bit_cast(float, x));
#else
    #pragma unroll
    for (int m = 1; m < 16; m <<= 1) t = fmaxf(t, __shfl_xor(t, m, 32));
#endif
    return t;
}

__global__ __launch_bounds__(NTHR) void fa_fwd_kernel(const float* __restrict__ Q,
                                                      const float* __restrict__ K,
                                                      const float* __restrict__ V,
                                                      float* __restrict__ Out)
{
    __shared__ _Float16 lK[BN * LSTR];           // K tile   [key][d]   f16
    __shared__ _Float16 lV[DDIM * LSTR];         // V tile   [d][key]   f16 (transposed)
    __shared__ _Float16 lP[NWAVE * 16 * LSTR];   // per-wave P [m][key] f16

    const int tid  = threadIdx.x;
    const int lane = tid & 31;
    const int wave = tid >> 5;
    const int hi   = lane >> 4;   // which half of the wave
    const int ln   = lane & 15;

    const int blk = blockIdx.x;   // query block index (BM rows)
    const int bh  = blockIdx.y;   // fused batch*head
    const int q0  = blk * BM + wave * 16;

    const size_t base  = (size_t)bh * TDIM * DDIM;
    // 1/sqrt(64) * log2(e): do the whole softmax in the base-2 domain
    const float  scale = 0.125f * 1.44269504088896340736f;

    // ---- Q A-fragments (16x32 f16, two D-chunks), scale folded in ----
    v16h qa[2];
    {
        const float* qrow = Q + base + (size_t)(q0 + ln) * DDIM;
        #pragma unroll
        for (int c = 0; c < 2; ++c) {
            const int d0 = c * 32 + hi * 8;
            #pragma unroll
            for (int i = 0; i < 8; i += 2) {
                pk2 a = __builtin_amdgcn_cvt_pkrtz(qrow[d0 + i] * scale,
                                                   qrow[d0 + i + 1] * scale);
                pk2 b = __builtin_amdgcn_cvt_pkrtz(qrow[d0 + 16 + i] * scale,
                                                   qrow[d0 + 16 + i + 1] * scale);
                qa[c][i]     = a[0]; qa[c][i + 1]     = a[1];
                qa[c][i + 8] = b[0]; qa[c][i + 8 + 1] = b[1];
            }
        }
    }

    // ones B-fragment: accumulates the softmax denominator with the same
    // recurrence as the output columns (one extra WMMA instead of a
    // cross-lane sum butterfly)
    v16h bones;
    #pragma unroll
    for (int i = 0; i < 16; ++i) bones[i] = (_Float16)1.0f;

    v8f   o[4], lacc = {};
    float mrun[8];
    #pragma unroll
    for (int c = 0; c < 4; ++c) o[c] = {};
    #pragma unroll
    for (int r = 0; r < 8; ++r) mrun[r] = -1e30f;

    _Float16* myP = &lP[wave * 16 * LSTR];

    const int ntiles = (blk * BM + BM + BN - 1) / BN;   // key tiles covering the block

    // ---- software pipeline: stage tile jt+1's global loads in registers
    //      while tile jt computes; single-buffered LDS ----
    float4 kreg[LITER], vreg[LITER];
    {
        const float* kb = K + base;
        const float* vb = V + base;
        #pragma unroll
        for (int it = 0; it < LITER; ++it) {
            const int elem = it * (NTHR * 4) + tid * 4;
            kreg[it] = *(const float4*)(kb + elem);
            vreg[it] = *(const float4*)(vb + elem);
        }
    }

    for (int jt = 0; jt < ntiles; ++jt) {
        // ---- drain staged registers into LDS (convert f32 -> f16 packed) ----
        #pragma unroll
        for (int it = 0; it < LITER; ++it) {
            const int elem = it * (NTHR * 4) + tid * 4;
            const int row  = elem >> 6;
            const int col  = elem & 63;
            pk2 k01 = __builtin_amdgcn_cvt_pkrtz(kreg[it].x, kreg[it].y);
            pk2 k23 = __builtin_amdgcn_cvt_pkrtz(kreg[it].z, kreg[it].w);
            h4 kh; kh[0] = k01[0]; kh[1] = k01[1]; kh[2] = k23[0]; kh[3] = k23[1];
            *(h4*)&lK[row * LSTR + col] = kh;
            pk2 v01 = __builtin_amdgcn_cvt_pkrtz(vreg[it].x, vreg[it].y);
            pk2 v23 = __builtin_amdgcn_cvt_pkrtz(vreg[it].z, vreg[it].w);
            lV[(col + 0) * LSTR + row] = v01[0];   // transposed store
            lV[(col + 1) * LSTR + row] = v01[1];
            lV[(col + 2) * LSTR + row] = v23[0];
            lV[(col + 3) * LSTR + row] = v23[1];
        }
        __syncthreads();

        // ---- issue next tile's global loads; they complete under the WMMAs ----
        if (jt + 1 < ntiles) {
            const float* kb = K + base + (size_t)(jt + 1) * BN * DDIM;
            const float* vb = V + base + (size_t)(jt + 1) * BN * DDIM;
            #pragma unroll
            for (int it = 0; it < LITER; ++it) {
                const int elem = it * (NTHR * 4) + tid * 4;
                kreg[it] = *(const float4*)(kb + elem);
                vreg[it] = *(const float4*)(vb + elem);
            }
        }

        // Wave-uniform guards: this wave's 16 rows are q0..q0+15.
        const bool active   = (jt * BN) <= (q0 + 15);        // tile has any key <= max row
        const bool needmask = (jt * BN + BN - 1) > q0;       // tile has any key > min row

        if (active) {
            // ---- S = Q * K^T : 4 key-chunks of 16, two WMMAs each over D ----
            v8f s[4];
            #pragma unroll
            for (int n = 0; n < 4; ++n) {
                v8f acc = {};
                #pragma unroll
                for (int c = 0; c < 2; ++c) {
                    const h8* bp = (const h8*)&lK[(n * 16 + ln) * LSTR + c * 32 + hi * 8];
                    h8 b0 = bp[0], b1 = bp[2];
                    v16h bf;
                    #pragma unroll
                    for (int i = 0; i < 8; ++i) { bf[i] = b0[i]; bf[i + 8] = b1[i]; }
                    acc = __builtin_amdgcn_wmma_f32_16x16x32_f16(
                            false, qa[c], false, bf, (short)0, acc, false, false);
                }
                s[n] = acc;
            }

            // ---- causal mask on the (partially) diagonal tile ----
            if (needmask) {
                #pragma unroll
                for (int n = 0; n < 4; ++n) {
                    const int key = jt * BN + n * 16 + ln;
                    #pragma unroll
                    for (int r = 0; r < 8; ++r) {
                        const int row = q0 + hi * 8 + r;
                        if (key > row) s[n][r] = -1e30f;
                    }
                }
            }

            // ---- online softmax in base-2 domain (row max via DPP butterfly) ----
            float alpha[8];
            #pragma unroll
            for (int r = 0; r < 8; ++r) {
                float t = fmaxf(fmaxf(s[0][r], s[1][r]), fmaxf(s[2][r], s[3][r]));
                t = red_max16(t);
                const float mnew = fmaxf(mrun[r], t);
                alpha[r] = __builtin_amdgcn_exp2f(mrun[r] - mnew);
                mrun[r]  = mnew;
            }
            #pragma unroll
            for (int n = 0; n < 4; ++n)
                #pragma unroll
                for (int r = 0; r < 8; ++r)
                    s[n][r] = __builtin_amdgcn_exp2f(s[n][r] - mrun[r]);
            #pragma unroll
            for (int c = 0; c < 4; ++c)
                #pragma unroll
                for (int r = 0; r < 8; ++r)
                    o[c][r] *= alpha[r];
            #pragma unroll
            for (int r = 0; r < 8; ++r)
                lacc[r] *= alpha[r];

            // ---- P (C-layout) -> per-wave LDS, f16 row-major ----
            #pragma unroll
            for (int n = 0; n < 4; ++n)
                #pragma unroll
                for (int r = 0; r < 8; r += 2) {
                    pk2 p = __builtin_amdgcn_cvt_pkrtz(s[n][r], s[n][r + 1]);
                    myP[(hi * 8 + r)     * LSTR + n * 16 + ln] = p[0];
                    myP[(hi * 8 + r + 1) * LSTR + n * 16 + ln] = p[1];
                }
            // LDS ops are in-order within a wave; region is wave-private -> no barrier

            // ---- O += P * V ; denominator += P * 1 ----
            #pragma unroll
            for (int c = 0; c < 2; ++c) {
                v16h pa;
                const h8* p0 = (const h8*)&myP[ln * LSTR + c * 32 + hi * 8];
                const h8* p1 = (const h8*)&myP[ln * LSTR + c * 32 + 16 + hi * 8];
                h8 a0 = p0[0], a1 = p1[0];
                #pragma unroll
                for (int i = 0; i < 8; ++i) { pa[i] = a0[i]; pa[i + 8] = a1[i]; }
                #pragma unroll
                for (int n = 0; n < 4; ++n) {
                    const h8* bp = (const h8*)&lV[(n * 16 + ln) * LSTR + c * 32 + hi * 8];
                    h8 b0 = bp[0], b1 = bp[2];
                    v16h bf;
                    #pragma unroll
                    for (int i = 0; i < 8; ++i) { bf[i] = b0[i]; bf[i + 8] = b1[i]; }
                    o[n] = __builtin_amdgcn_wmma_f32_16x16x32_f16(
                            false, pa, false, bf, (short)0, o[n], false, false);
                }
                lacc = __builtin_amdgcn_wmma_f32_16x16x32_f16(
                        false, pa, false, bones, (short)0, lacc, false, false);
            }
        }

        __syncthreads();  // all waves done reading lK/lV before next store phase
    }

    // ---- epilogue: normalize (denominator is replicated across lanes) ----
    float* ob = Out + base + (size_t)q0 * DDIM;
    float inv[8];
    #pragma unroll
    for (int r = 0; r < 8; ++r) inv[r] = __builtin_amdgcn_rcpf(lacc[r]);
    #pragma unroll
    for (int c = 0; c < 4; ++c) {
        #pragma unroll
        for (int r = 0; r < 8; ++r) {
            ob[(hi * 8 + r) * DDIM + c * 16 + ln] = o[c][r] * inv[r];
        }
    }
}

extern "C" void kernel_launch(void* const* d_in, const int* in_sizes, int n_in,
                              void* d_out, int out_size, void* d_ws, size_t ws_size,
                              hipStream_t stream) {
    (void)in_sizes; (void)n_in; (void)out_size; (void)d_ws; (void)ws_size;
    const float* q = (const float*)d_in[0];
    const float* k = (const float*)d_in[1];
    const float* v = (const float*)d_in[2];
    float* out = (float*)d_out;
    dim3 grid(TDIM / BM, BHDIM, 1);   // 16 query blocks x 32 (batch*head)
    fa_fwd_kernel<<<grid, NTHR, 0, stream>>>(q, k, v, out);
}
